// TransformerBlock_30502857736559
// MI455X (gfx1250) — matrix operand order repaired and checked by reading, not compiled, not run on hardware
//
#include <hip/hip_runtime.h>
#include <hip/hip_bf16.h>
#include <math.h>

// ---------------------------------------------------------------------------
// TransformerBlock for MI455X (gfx1250), wave32 + v_wmma_f32_16x16x32_bf16.
// Compute-bound (≈275 GFLOP vs ≈250MB traffic @23.3TB/s) => all GEMMs on WMMA.
// Attention: flash-style online softmax in exp2 domain; row-sum via P@ones WMMA;
// row-max via DPP16 xor reduction (pure VALU, no LDS round-trips).
// ---------------------------------------------------------------------------

typedef __bf16 bf16;
typedef __attribute__((ext_vector_type(16))) bf16  v16bf;
typedef __attribute__((ext_vector_type(8)))  bf16  v8bf;
typedef __attribute__((ext_vector_type(8)))  float v8f;

#define DMODEL 1024
#define DHID   4096
#define SEQ    2048
#define BATCH  4
#define NHEAD  16
#define DKH    64
#define ROWS   (BATCH * SEQ)   // 8192

// 1/sqrt(DKH) * log2(e): softmax computed in base-2 domain (v_exp_f32 native)
#define SCORE_SCALE_LOG2 0.18033688011112042592f

// ---------------------------------------------------------------------------
// Fragment loaders (contiguous k order, same for A and B; not the ISA table order; wave32)
// A (bf16 16x32): lane L<16 holds row M=L, K = 0..15;
//                 lane L>=16 holds row M=L-16, K = 16..31.
__device__ __forceinline__ v16bf load_a_frag(const bf16* __restrict__ A, int lda,
                                             int m_base, int k0, int lane) {
    const bf16* p = A + (size_t)(m_base + (lane & 15)) * lda + k0 + ((lane >> 4) << 4);
    v8bf lo = *(const v8bf*)p;          // K = g*16 + 0..7
    v8bf hi = *(const v8bf*)(p + 8);    // K = g*16 + 8..15
    return __builtin_shufflevector(lo, hi, 0,1,2,3,4,5,6,7,8,9,10,11,12,13,14,15);
}
// B (bf16 32x16): lane L holds column N=L&15; K = (L>>4)*16 + 0..15 (contiguous
// in memory because weights are [out,in] row-major => contiguous along K).
__device__ __forceinline__ v16bf load_b_frag(const bf16* __restrict__ Bw, int ldb,
                                             int n_base, int k0, int lane) {
    const bf16* p = Bw + (size_t)(n_base + (lane & 15)) * ldb + k0 + ((lane >> 4) << 4);
    return *(const v16bf*)p;
}

__device__ __forceinline__ v8f wmma_bf16(v16bf a, v16bf b, v8f c) {
    return __builtin_amdgcn_wmma_f32_16x16x32_bf16(false, a, false, b,
                                                   (short)0, c, false, false);
}

// 16-lane max reduction using DPP16 row ops (stays inside each row of 16):
// quad_perm xor1, quad_perm xor2, row_half_mirror (^7), row_mirror (^15).
template <int CTRL>
__device__ __forceinline__ float max_dpp_step(float v) {
    const int s = __builtin_amdgcn_update_dpp(0, __float_as_int(v),
                                              CTRL, 0xF, 0xF, true);
    return fmaxf(v, __int_as_float(s));
}
__device__ __forceinline__ float maxred16_dpp(float v) {
    v = max_dpp_step<0xB1>(v);   // quad_perm(1,0,3,2): xor 1
    v = max_dpp_step<0x4E>(v);   // quad_perm(2,3,0,1): xor 2
    v = max_dpp_step<0x141>(v);  // ROW_HALF_MIRROR:    xor within 8
    v = max_dpp_step<0x140>(v);  // ROW_MIRROR:         xor within 16
    return v;
}

// ---------------------------------------------------------------------------
// Generic bf16 GEMM: out[M,N] = A[M,K] @ Bw[N,K]^T  (+ epilogue)
// Block = 256 threads = 8 waves (2 in M x 4 in N); wave tile 64x32; block 128x128.
// EPI: 0 = store bf16; 1 = +bias, exact GELU, store bf16;
//      2 = +residual, store f32; 3 = +bias +residual, store f32.
template <int EPI>
__global__ __launch_bounds__(256) void gemm_bf16_kernel(
    const bf16* __restrict__ A, const bf16* __restrict__ Bw,
    int M, int N, int K,
    bf16* __restrict__ outB, float* __restrict__ outF,
    const float* __restrict__ bias, const float* __restrict__ residual)
{
    const int lane = threadIdx.x & 31;
    const int wave = threadIdx.x >> 5;
    const int m0 = blockIdx.y * 128 + (wave >> 2) * 64;
    const int n0 = blockIdx.x * 128 + (wave & 3) * 32;

    v8f zero = {0.f,0.f,0.f,0.f,0.f,0.f,0.f,0.f};
    v8f acc[4][2];
    #pragma unroll
    for (int t = 0; t < 4; ++t) { acc[t][0] = zero; acc[t][1] = zero; }

    for (int k0 = 0; k0 < K; k0 += 32) {
        // prefetch next weight chunk into cache (global_prefetch_b8)
        if (k0 + 32 < K)
            __builtin_prefetch(Bw + (size_t)(n0 + (lane & 15)) * K + k0 + 32, 0, 3);
        v16bf bfr0 = load_b_frag(Bw, K, n0,      k0, lane);
        v16bf bfr1 = load_b_frag(Bw, K, n0 + 16, k0, lane);
        #pragma unroll
        for (int t = 0; t < 4; ++t) {
            v16bf afr = load_a_frag(A, K, m0 + t * 16, k0, lane);
            acc[t][0] = wmma_bf16(afr, bfr0, acc[t][0]);
            acc[t][1] = wmma_bf16(afr, bfr1, acc[t][1]);
        }
    }

    // C layout: VGPR r -> row m0+t*16+r (lanes 0..15) / +8 (lanes 16..31), col = lane&15.
    const int ng = lane & 15;
    const int mg = (lane >> 4) * 8;
    #pragma unroll
    for (int s = 0; s < 2; ++s) {
        const int n = n0 + s * 16 + ng;
        float bv = 0.f;
        if (EPI == 1 || EPI == 3) bv = bias[n];
        #pragma unroll
        for (int t = 0; t < 4; ++t) {
            #pragma unroll
            for (int r = 0; r < 8; ++r) {
                const int m = m0 + t * 16 + mg + r;
                const size_t idx = (size_t)m * N + n;
                float v = acc[t][s][r];
                if (EPI == 0) {
                    outB[idx] = (bf16)v;
                } else if (EPI == 1) {
                    v += bv;
                    v = 0.5f * v * (1.f + erff(v * 0.70710678118654752f));
                    outB[idx] = (bf16)v;
                } else if (EPI == 2) {
                    outF[idx] = v + residual[idx];
                } else {
                    outF[idx] = v + bv + residual[idx];
                }
            }
        }
    }
}

// ---------------------------------------------------------------------------
// LayerNorm over D=1024, one block (256 threads) per row, output bf16.
__global__ __launch_bounds__(256) void ln_kernel(
    const float* __restrict__ x, const float* __restrict__ w,
    const float* __restrict__ b, bf16* __restrict__ out)
{
    __shared__ float red[256];
    const int row = blockIdx.x;
    const int tid = threadIdx.x;
    const float* xr = x + (size_t)row * DMODEL;
    float v[4];
    float s = 0.f;
    #pragma unroll
    for (int i = 0; i < 4; ++i) { v[i] = xr[tid + i * 256]; s += v[i]; }
    red[tid] = s; __syncthreads();
    for (int off = 128; off > 0; off >>= 1) {
        if (tid < off) red[tid] += red[tid + off];
        __syncthreads();
    }
    const float mean = red[0] * (1.f / DMODEL);
    __syncthreads();
    s = 0.f;
    #pragma unroll
    for (int i = 0; i < 4; ++i) { float d = v[i] - mean; s += d * d; }
    red[tid] = s; __syncthreads();
    for (int off = 128; off > 0; off >>= 1) {
        if (tid < off) red[tid] += red[tid + off];
        __syncthreads();
    }
    const float rstd = rsqrtf(red[0] * (1.f / DMODEL) + 1e-5f);
    #pragma unroll
    for (int i = 0; i < 4; ++i) {
        const int col = tid + i * 256;
        out[(size_t)row * DMODEL + col] = (bf16)((v[i] - mean) * rstd * w[col] + b[col]);
    }
}

// ---------------------------------------------------------------------------
// f32 -> bf16 cast (weights)
__global__ __launch_bounds__(256) void cast_bf16_kernel(
    const float* __restrict__ in, bf16* __restrict__ out, int n)
{
    const int i = blockIdx.x * 256 + threadIdx.x;
    if (i < n) out[i] = (bf16)in[i];
}

// ---------------------------------------------------------------------------
// V slice of bf16 QKV -> vT[b,h,dk,n]  (so P*V B-fragments are contiguous per lane)
__global__ __launch_bounds__(256) void v_transpose_kernel(
    const bf16* __restrict__ qkv, bf16* __restrict__ vT)
{
    const int i = blockIdx.x * 256 + threadIdx.x;   // over [bh][dk][n]
    if (i >= BATCH * NHEAD * DKH * SEQ) return;
    const int n   = i & (SEQ - 1);
    const int r1  = i >> 11;
    const int dk  = r1 & (DKH - 1);
    const int bh  = r1 >> 6;
    const int b   = bh >> 4;
    const int h   = bh & 15;
    vT[i] = qkv[((size_t)(b * SEQ + n)) * (3 * DMODEL) + 2 * DMODEL + h * DKH + dk];
}

// ---------------------------------------------------------------------------
// Flash attention: one wave = 16 query rows of one (b,h); online softmax over
// 32-key chunks in exp2 domain. Row sums via an extra P@ones WMMA (each column
// of that C tile replicates the row sum in exactly the per-lane layout of the
// softmax state). Row max via DPP16 xor reduction (VALU-only, waitless).
__global__ __launch_bounds__(256) void attn_kernel(
    const bf16* __restrict__ qkv,   // [B*SEQ, 3*DMODEL]
    const bf16* __restrict__ vT,    // [B*H, DKH, SEQ]
    bf16* __restrict__ o)           // [B*SEQ, DMODEL]
{
    __shared__ bf16 pbuf[8][16 * 32];
    const int lane = threadIdx.x & 31;
    const int wave = threadIdx.x >> 5;
    const int task = blockIdx.x * 8 + wave;         // 8192 tasks
    const int ntile = SEQ / 16;                     // 128
    const int bh = task / ntile;
    const int qt = task % ntile;
    const int b = bh >> 4, h = bh & 15;
    const int row_g = lane & 15;
    const int grp   = lane >> 4;
    const int n0 = qt * 16;

    // Q fragments (A layout), K-dim = head dim 64 -> two K=32 halves.
    const bf16* qbase = qkv + ((size_t)(b * SEQ + n0 + row_g)) * (3 * DMODEL) + h * DKH;
    v16bf qf[2];
    #pragma unroll
    for (int half = 0; half < 2; ++half) {
        const bf16* p = qbase + half * 32 + grp * 16;
        v8bf lo = *(const v8bf*)p;
        v8bf hi = *(const v8bf*)(p + 8);
        qf[half] = __builtin_shufflevector(lo, hi, 0,1,2,3,4,5,6,7,8,9,10,11,12,13,14,15);
    }
    const bf16* kbase = qkv + (size_t)(b * SEQ) * (3 * DMODEL) + DMODEL + h * DKH;
    const bf16* vbase = vT + (size_t)bh * DKH * SEQ;

    // all-ones B fragment for the row-sum WMMA
    v16bf vones;
    #pragma unroll
    for (int i = 0; i < 16; ++i) vones[i] = (bf16)1.0f;

    float Mrow[8];
    v8f zero = {0.f,0.f,0.f,0.f,0.f,0.f,0.f,0.f};
    v8f oacc[4];
    v8f lacc = zero;                 // running softmax denominator per row
    #pragma unroll
    for (int r = 0; r < 8; ++r) Mrow[r] = -INFINITY;
    #pragma unroll
    for (int t = 0; t < 4; ++t) oacc[t] = zero;

    for (int c = 0; c < SEQ; c += 32) {
        // S = Q K^T (raw) for keys [c, c+32); scale folded in later via fma.
        v8f s0 = zero, s1 = zero;
        #pragma unroll
        for (int half = 0; half < 2; ++half) {
            // B fragment: col = key tile col, K = d (contiguous along head dim)
            v16bf kf0 = *(const v16bf*)(kbase + (size_t)(c + row_g)      * (3 * DMODEL) + half * 32 + grp * 16);
            v16bf kf1 = *(const v16bf*)(kbase + (size_t)(c + 16 + row_g) * (3 * DMODEL) + half * 32 + grp * 16);
            s0 = wmma_bf16(qf[half], kf0, s0);
            s1 = wmma_bf16(qf[half], kf1, s1);
        }
        // Row max on raw scores: DPP16 xor reduction per row (no LDS, no waits).
        float mx[8];
        #pragma unroll
        for (int r = 0; r < 8; ++r) mx[r] = maxred16_dpp(fmaxf(s0[r], s1[r]));
        // Online-softmax state update (base-2 domain).
        #pragma unroll
        for (int r = 0; r < 8; ++r) {
            const float newM = fmaxf(Mrow[r], mx[r] * SCORE_SCALE_LOG2);
            const float corr = __builtin_amdgcn_exp2f(Mrow[r] - newM);  // v_exp_f32
            Mrow[r] = newM;
            lacc[r] *= corr;
            #pragma unroll
            for (int t = 0; t < 4; ++t) oacc[t][r] *= corr;
        }
        // P = exp2(s*scale - M) via fma; stage to LDS (C layout -> A layout).
        bf16* pb = pbuf[wave];
        #pragma unroll
        for (int r = 0; r < 8; ++r) {
            const float p0 = __builtin_amdgcn_exp2f(fmaf(s0[r], SCORE_SCALE_LOG2, -Mrow[r]));
            const float p1 = __builtin_amdgcn_exp2f(fmaf(s1[r], SCORE_SCALE_LOG2, -Mrow[r]));
            const int rr = r + grp * 8;
            pb[rr * 32 + row_g]      = (bf16)p0;
            pb[rr * 32 + 16 + row_g] = (bf16)p1;
        }
        // P as A fragment (same-wave LDS RAW; compiler inserts s_wait_dscnt)
        const bf16* pp = pb + row_g * 32 + grp * 16;
        v8bf plo = *(const v8bf*)pp;
        v8bf phi = *(const v8bf*)(pp + 8);
        v16bf pf = __builtin_shufflevector(plo, phi, 0,1,2,3,4,5,6,7,8,9,10,11,12,13,14,15);
        // l += P @ ones  (row sums, replicated across lanes of the group)
        lacc = wmma_bf16(pf, vones, lacc);
        // O += P @ V   (V^T fragments contiguous per lane)
        #pragma unroll
        for (int t = 0; t < 4; ++t) {
            v16bf vf = *(const v16bf*)(vbase + (size_t)(t * 16 + row_g) * SEQ + c + grp * 16);
            oacc[t] = wmma_bf16(pf, vf, oacc[t]);
        }
    }
    // normalize and write [b, n, h*64 + dk] as bf16
    #pragma unroll
    for (int r = 0; r < 8; ++r) {
        const float inv = 1.f / lacc[r];
        const int m = n0 + grp * 8 + r;
        const size_t orow = ((size_t)(b * SEQ + m)) * DMODEL + h * DKH;
        #pragma unroll
        for (int t = 0; t < 4; ++t)
            o[orow + t * 16 + row_g] = (bf16)(oacc[t][r] * inv);
    }
}

// ---------------------------------------------------------------------------
extern "C" void kernel_launch(void* const* d_in, const int* in_sizes, int n_in,
                              void* d_out, int out_size, void* d_ws, size_t ws_size,
                              hipStream_t stream) {
    (void)in_sizes; (void)n_in; (void)out_size; (void)ws_size;
    const float* x      = (const float*)d_in[0];
    const float* ln1_w  = (const float*)d_in[1];
    const float* ln1_b  = (const float*)d_in[2];
    const float* w_qkv  = (const float*)d_in[3];
    const float* w_out  = (const float*)d_in[4];
    const float* ln2_w  = (const float*)d_in[5];
    const float* ln2_b  = (const float*)d_in[6];
    const float* w_fc1  = (const float*)d_in[7];
    const float* b_fc1  = (const float*)d_in[8];
    const float* w_fc2  = (const float*)d_in[9];
    const float* b_fc2  = (const float*)d_in[10];
    float* out = (float*)d_out;

    // workspace carve-up (≈244 MB total)
    char* p = (char*)d_ws;
    auto alloc = [&](size_t bytes) -> char* {
        char* r = p; p += (bytes + 255) & ~(size_t)255; return r;
    };
    bf16*  x_ln   = (bf16*)alloc((size_t)ROWS * DMODEL * 2);
    bf16*  wqkv_b = (bf16*)alloc((size_t)3 * DMODEL * DMODEL * 2);
    bf16*  wout_b = (bf16*)alloc((size_t)DMODEL * DMODEL * 2);
    bf16*  wfc1_b = (bf16*)alloc((size_t)DHID * DMODEL * 2);
    bf16*  wfc2_b = (bf16*)alloc((size_t)DMODEL * DHID * 2);
    bf16*  qkv_b  = (bf16*)alloc((size_t)ROWS * 3 * DMODEL * 2);
    bf16*  vT_b   = (bf16*)alloc((size_t)ROWS * DMODEL * 2);
    bf16*  ao_b   = (bf16*)alloc((size_t)ROWS * DMODEL * 2);
    float* x2     = (float*)alloc((size_t)ROWS * DMODEL * 4);
    bf16*  h2_ln  = (bf16*)alloc((size_t)ROWS * DMODEL * 2);
    bf16*  act_b  = (bf16*)alloc((size_t)ROWS * DHID * 2);

    // 1) weights -> bf16
    auto cast = [&](const float* src, bf16* dst, int n) {
        cast_bf16_kernel<<<(n + 255) / 256, 256, 0, stream>>>(src, dst, n);
    };
    cast(w_qkv, wqkv_b, 3 * DMODEL * DMODEL);
    cast(w_out, wout_b, DMODEL * DMODEL);
    cast(w_fc1, wfc1_b, DHID * DMODEL);
    cast(w_fc2, wfc2_b, DMODEL * DHID);

    // 2) LN1
    ln_kernel<<<ROWS, 256, 0, stream>>>(x, ln1_w, ln1_b, x_ln);

    // 3) QKV = x_ln @ w_qkv^T  -> bf16 [ROWS, 3D]
    gemm_bf16_kernel<0><<<dim3(3 * DMODEL / 128, ROWS / 128), 256, 0, stream>>>(
        x_ln, wqkv_b, ROWS, 3 * DMODEL, DMODEL, qkv_b, nullptr, nullptr, nullptr);

    // 4) transpose V -> [B*H, DK, SEQ]
    v_transpose_kernel<<<(BATCH * NHEAD * DKH * SEQ) / 256, 256, 0, stream>>>(qkv_b, vT_b);

    // 5) flash attention -> ao_b [ROWS, D]
    attn_kernel<<<(BATCH * NHEAD * (SEQ / 16)) / 8, 256, 0, stream>>>(qkv_b, vT_b, ao_b);

    // 6) x2 = x + ao @ w_out^T   (f32)
    gemm_bf16_kernel<2><<<dim3(DMODEL / 128, ROWS / 128), 256, 0, stream>>>(
        ao_b, wout_b, ROWS, DMODEL, DMODEL, nullptr, x2, nullptr, x);

    // 7) LN2
    ln_kernel<<<ROWS, 256, 0, stream>>>(x2, ln2_w, ln2_b, h2_ln);

    // 8) act = gelu(h2 @ w_fc1^T + b_fc1) -> bf16 [ROWS, DHID]
    gemm_bf16_kernel<1><<<dim3(DHID / 128, ROWS / 128), 256, 0, stream>>>(
        h2_ln, wfc1_b, ROWS, DHID, DMODEL, act_b, nullptr, b_fc1, nullptr);

    // 9) out = x2 + act @ w_fc2^T + b_fc2  (f32)
    gemm_bf16_kernel<3><<<dim3(DMODEL / 128, ROWS / 128), 256, 0, stream>>>(
        act_b, wfc2_b, ROWS, DMODEL, DHID, nullptr, out, b_fc2, x2);
}
